// LSTM_78005196029961
// MI455X (gfx1250) — matrix operand order repaired and checked by reading, not compiled, not run on hardware
//
#include <hip/hip_runtime.h>
#include <math.h>

// Problem constants (match reference)
#define HDIM  64
#define TLEN  32000
#define BATCH 16
#define XCH   64     // x-staging chunk (timesteps)

typedef _Float16 v16h __attribute__((ext_vector_type(16)));
typedef float    v8f  __attribute__((ext_vector_type(8)));

union AView { v16h v; uint4 q[2]; };

// ---- CDNA5 fast activations -------------------------------------------------
// gfx1250 has a single-instruction v_tanh_f32 (TRANS unit). Use it when the
// builtin exists; otherwise fall back to exp + v_rcp_f32 (no IEEE-div Newton
// chains — __builtin_amdgcn_rcpf is a single TRANS op and is far more accurate
// than our f16 matmul inputs, so precision is unaffected).
#if __has_builtin(__builtin_amdgcn_tanhf)
__device__ __forceinline__ float tanh_f(float x) {
    return __builtin_amdgcn_tanhf(x);
}
__device__ __forceinline__ float sigmoid_f(float x) {
    // sigmoid(x) = 0.5*tanh(x/2) + 0.5   -> mul + v_tanh + fma
    return __builtin_fmaf(0.5f, __builtin_amdgcn_tanhf(0.5f * x), 0.5f);
}
#else
__device__ __forceinline__ float tanh_f(float x) {
    // stable: tanh(x) = 1 - 2/(exp(2x)+1)
    float e = __expf(2.0f * x);
    return __builtin_fmaf(-2.0f, __builtin_amdgcn_rcpf(e + 1.0f), 1.0f);
}
__device__ __forceinline__ float sigmoid_f(float x) {
    return __builtin_amdgcn_rcpf(1.0f + __expf(-x));
}
#endif

// One workgroup per LSTM direction. 4 waves (wave32) = 4 SIMDs of one WGP.
// Wave j owns gate-columns [16j,16j+16) of each of the 4 gates, so the
// c/h update is register-resident; only h crosses LDS (double-buffered,
// ONE barrier per timestep on the 32000-long dependence chain).
__global__ __launch_bounds__(128)
void lstm_bidir_persistent(const float* __restrict__ x,
                           const float* __restrict__ w_ih_f, const float* __restrict__ w_hh_f,
                           const float* __restrict__ b_ih_f, const float* __restrict__ b_hh_f,
                           const float* __restrict__ w_ih_b, const float* __restrict__ w_hh_b,
                           const float* __restrict__ b_ih_b, const float* __restrict__ b_hh_b,
                           const float* __restrict__ w_head,
                           float* __restrict__ acc /* [BATCH] partial logits in ws */)
{
    const int dir  = blockIdx.x;        // 0 = forward, 1 = backward
    const int tid  = threadIdx.x;
    const int j    = tid >> 5;          // wave id 0..3
    const int lane = tid & 31;
    const int llo  = lane & 15;         // tile column / A row (M)
    const int lhi  = lane >> 4;         // K-half selector per ISA layouts

    const float* w_ih = dir ? w_ih_b : w_ih_f;
    const float* w_hh = dir ? w_hh_b : w_hh_f;
    const float* b_ih = dir ? b_ih_b : b_ih_f;
    const float* b_hh = dir ? b_hh_b : b_hh_f;

    __shared__ _Float16 h_lds[2][BATCH][HDIM];   // double-buffered hidden state (f16)
    __shared__ float    x_lds[XCH][BATCH];       // staged input chunk
    __shared__ float    red[4][BATCH];           // head reduction

    // ---- Preload W_hh^T tiles into VGPRs as f16 in the ISA B-matrix layout.
    // B is [K=64][N]; column N = gate*64 + 16j + llo = W_hh row g.
    // v16h element e of K-block kb holds k = kb*32 + 8*lhi + e (e<8), kb*32 + 16 + 8*lhi + (e-8) (e>=8).
    v16h Bw[4][2];
    float biasc[4], wihc[4];
    #pragma unroll
    for (int gate = 0; gate < 4; ++gate) {
        const int g = gate * 64 + j * 16 + llo;        // gate order i,f,g,o
        biasc[gate] = b_ih[g] + b_hh[g];
        wihc[gate]  = w_ih[g];                          // input_size == 1
        #pragma unroll
        for (int kb = 0; kb < 2; ++kb) {
            v16h bt;
            #pragma unroll
            for (int e = 0; e < 16; ++e) {
                const int k = kb * 32 + (e < 8 ? 8 * lhi + e : 16 + 8 * lhi + (e - 8));
                bt[e] = (_Float16)w_hh[g * HDIM + k];
            }
            Bw[gate][kb] = bt;
        }
    }

    // h0 = 0 in read-buffer 0
    for (int e = tid; e < BATCH * HDIM; e += 128)
        ((_Float16*)h_lds[0])[e] = (_Float16)0.0f;
    __syncthreads();

    float creg[8], headacc[8];
    #pragma unroll
    for (int v = 0; v < 8; ++v) { creg[v] = 0.0f; headacc[v] = 0.0f; }

    for (int s = 0; s < TLEN; ++s) {
        const int t = dir ? (TLEN - 1 - s) : s;

        // ---- stage next x chunk (previous step's barrier fenced old readers)
        if ((s & (XCH - 1)) == 0) {
            #pragma unroll
            for (int r = 0; r < 8; ++r) {
                const int e  = r * 128 + tid;          // 0..1023
                const int si = e >> 4, b = e & 15;
                const int tt = dir ? (TLEN - 1 - (s + si)) : (s + si);
                x_lds[si][b] = x[b * TLEN + tt];
            }
            __syncthreads();
        }

        const int rb = s & 1, wb = rb ^ 1;

        // ---- A-matrix (h, f16) from LDS in ISA 16-bit A layout:
        // lane row M=llo; elements 0..7 -> K = 8*lhi..8*lhi+7, 8..15 -> +16.
        const _Float16* hrow = &h_lds[rb][llo][0];
        AView A0, A1;
        A0.q[0] = *(const uint4*)(hrow +      8 * lhi);
        A0.q[1] = *(const uint4*)(hrow + 16 + 8 * lhi);
        A1.q[0] = *(const uint4*)(hrow + 32 + 8 * lhi);
        A1.q[1] = *(const uint4*)(hrow + 48 + 8 * lhi);

        // per-row x values (C-tile row b = v + 8*lhi)
        const float4 xq0 = *(const float4*)&x_lds[s & (XCH - 1)][8 * lhi + 0];
        const float4 xq1 = *(const float4*)&x_lds[s & (XCH - 1)][8 * lhi + 4];
        const float xv[8] = { xq0.x, xq0.y, xq0.z, xq0.w, xq1.x, xq1.y, xq1.z, xq1.w };

        // ---- gates = (x*w_ih + bias) + h @ W_hh^T   via 2 chained WMMAs/gate
        float gv[4][8];
        #pragma unroll
        for (int gate = 0; gate < 4; ++gate) {
            v8f cacc;
            #pragma unroll
            for (int v = 0; v < 8; ++v) cacc[v] = __builtin_fmaf(xv[v], wihc[gate], biasc[gate]);
            cacc = __builtin_amdgcn_wmma_f32_16x16x32_f16(false, A0.v, false, Bw[gate][0],
                                                          (short)0, cacc, false, false);
            cacc = __builtin_amdgcn_wmma_f32_16x16x32_f16(false, A1.v, false, Bw[gate][1],
                                                          (short)0, cacc, false, false);
            #pragma unroll
            for (int v = 0; v < 8; ++v) gv[gate][v] = cacc[v];
        }

        // ---- activations, state update, fused head dot, h -> LDS (f16)
        const float wmul = w_head[t * (2 * HDIM) + dir * HDIM + j * 16 + llo];
        #pragma unroll
        for (int v = 0; v < 8; ++v) {
            const float iv = sigmoid_f(gv[0][v]);
            const float fv = sigmoid_f(gv[1][v]);
            const float gg = tanh_f   (gv[2][v]);
            const float ov = sigmoid_f(gv[3][v]);
            const float c  = __builtin_fmaf(fv, creg[v], iv * gg);
            creg[v] = c;
            const float hv = ov * tanh_f(c);
            headacc[v] = __builtin_fmaf(hv, wmul, headacc[v]);
            h_lds[wb][v + 8 * lhi][j * 16 + llo] = (_Float16)hv;
        }
        __syncthreads();   // the single per-step barrier
    }

    // ---- head reduction: sum over the 16 columns each half-wave holds
    #pragma unroll
    for (int v = 0; v < 8; ++v) {
        float sum = headacc[v];
        #pragma unroll
        for (int m = 1; m < 16; m <<= 1)
            sum += __shfl_xor(sum, m, 32);
        if (llo == 0) red[j][8 * lhi + v] = sum;   // batch b = v + 8*lhi
    }
    __syncthreads();
    if (tid < BATCH) {
        const float sb = red[0][tid] + red[1][tid] + red[2][tid] + red[3][tid];
        atomicAdd(&acc[tid], sb);
    }
}

__global__ void zero_acc_kernel(float* acc) {
    if (threadIdx.x < BATCH) acc[threadIdx.x] = 0.0f;
}

__global__ void finalize_kernel(const float* __restrict__ acc,
                                const float* __restrict__ b_head,
                                float* __restrict__ out) {
    const int b = threadIdx.x;
    if (b < BATCH) out[b] = 1.0f / (1.0f + __expf(-(acc[b] + b_head[0])));
}

extern "C" void kernel_launch(void* const* d_in, const int* in_sizes, int n_in,
                              void* d_out, int out_size, void* d_ws, size_t ws_size,
                              hipStream_t stream) {
    (void)in_sizes; (void)n_in; (void)out_size; (void)ws_size;
    const float* x      = (const float*)d_in[0];
    const float* w_ih_f = (const float*)d_in[1];
    const float* w_hh_f = (const float*)d_in[2];
    const float* b_ih_f = (const float*)d_in[3];
    const float* b_hh_f = (const float*)d_in[4];
    const float* w_ih_b = (const float*)d_in[5];
    const float* w_hh_b = (const float*)d_in[6];
    const float* b_ih_b = (const float*)d_in[7];
    const float* b_hh_b = (const float*)d_in[8];
    const float* w_head = (const float*)d_in[9];
    const float* b_head = (const float*)d_in[10];

    float* acc = (float*)d_ws;   // 16 floats of scratch

    zero_acc_kernel<<<1, 32, 0, stream>>>(acc);
    lstm_bidir_persistent<<<2, 128, 0, stream>>>(x, w_ih_f, w_hh_f, b_ih_f, b_hh_f,
                                                 w_ih_b, w_hh_b, b_ih_b, b_hh_b,
                                                 w_head, acc);
    finalize_kernel<<<1, 32, 0, stream>>>(acc, b_head, (float*)d_out);
}
